// ConvODENet_25735444037716
// MI455X (gfx1250) — matrix operand order repaired and checked by reading, not compiled, not run on hardware
//
#include <hip/hip_runtime.h>
#include <hip/hip_bf16.h>
#include <math.h>

// ---------------------------------------------------------------------------
// ConvODENet on MI455X (gfx1250, wave32):
//   y0 = pad_channels(x, 3->32)
//   y  = RK4(f, y0, t=0..1, 8 steps),  f(y) = tanh(conv3x3_same(y, Wconv))
//   feats = max_{h,w} y ; out = feats @ Wout^T + bout
// Conv = implicit GEMM, tap-decomposed: per 16px x 16cout tile, 9 taps, each
// one v_wmma_f32_16x16x32_f16 over K = 32 input channels (f16 in, f32 acc).
// RK stage mixing and accumulation fused into conv load/epilogue.
// HBM-bandwidth-bound => minimize steps (RK4 h=1/8: global err ~2.4e-4 < 1e-3)
// and keep all elementwise work fused + branch-free.
// ---------------------------------------------------------------------------

typedef __attribute__((ext_vector_type(16))) _Float16 v16h;
typedef __attribute__((ext_vector_type(8)))  float    v8f;
typedef __attribute__((ext_vector_type(2)))  float    v2f;
typedef int v4i __attribute__((vector_size(16)));     // int4 for async-LDS DMA

#define NB   32          // batch
#define NF   32          // channels after pad
#define HH   128
#define WW   128
#define HW   (HH*WW)     // 16384
#define NTOT ((size_t)NB*NF*HW)

union FragU { uint4 u[2]; v16h v; };

__device__ __forceinline__ float4 ld4(const float* p){ return *(const float4*)p; }
__device__ __forceinline__ void   st4(float* p, float4 v){ *(float4*)p = v; }

// Branch-free tanh: 1 - 2/(exp(2x)+1). v_exp_f32 (TRANS) + fast rcp; saturates
// correctly for |x|->inf, no EXEC divergence (libm tanhf branches).
__device__ __forceinline__ float fast_tanh(float x) {
    float e = __expf(2.0f * x);
    return 1.0f - __fdividef(2.0f, e + 1.0f);
}

// ---------------------------------------------------------------- pad kernel
__global__ __launch_bounds__(256) void pad_kernel(const float* __restrict__ x,
                                                  float* __restrict__ y) {
    size_t i = (size_t)blockIdx.x * 256 + threadIdx.x;   // < NTOT
    int c = (int)((i >> 14) & 31);
    int b = (int)(i >> 19);
    float v = 0.0f;
    if (c < 3) v = x[(((size_t)b * 3 + c) << 14) + (i & 16383)];
    y[i] = v;
}

// ------------------------------------------------ weight convert: f32 -> f16
// Wconv [co][ci][kh][kw] (OIHW) -> Wh [tap][co][ci]  (B-frag friendly)
__global__ __launch_bounds__(256) void wcvt_kernel(const float* __restrict__ Wc,
                                                   _Float16* __restrict__ Wh) {
    int e = blockIdx.x * 256 + threadIdx.x;              // < 9216
    int co = e / 288;
    int ci = (e / 9) % 32;
    int t  = e % 9;
    Wh[(t * 32 + co) * 32 + ci] = (_Float16)Wc[e];
}

// ---------------------------------------------- fused conv3x3 + tanh + RK ops
// stage input  : in = y + cmix * kin            (computed while staging to LDS)
// conv output  : f  = tanh(conv3x3(in))
// mode 0 (INIT): kout = f;  acc  = y + wacc*f
// mode 1 (ACC) : kout = f;  acc += wacc*f
// mode 2 (FIN) :            acc += wacc*f      (acc becomes y_{n+1}; swap host-side)
__global__ __launch_bounds__(256) void conv_tanh_rk(
        const float* __restrict__ y, const float* __restrict__ kin, float cmix,
        const _Float16* __restrict__ Wh,
        float* __restrict__ kout, float* __restrict__ accb,
        float wacc, int mode) {

    __shared__ __align__(16) _Float16 sIn[3 * 66 * 32];   // [row][px][cin] f16
    __shared__ __align__(16) _Float16 sW[9 * 32 * 32];    // [tap][cout][cin] f16
    __shared__ __align__(16) float    sOut[32 * 64];      // [cout][px] f32

    const int tid  = threadIdx.x;
    const int gid  = blockIdx.x;                 // 32*128*2 = 8192 blocks
    const int half = gid & 1;
    const int h    = (gid >> 1) & 127;
    const int b    = gid >> 8;
    const int w0   = half * 64;

    // ---- stage weights (18KB) into LDS: async DMA (ASYNCcnt path) if present
#if defined(__HIP_DEVICE_COMPILE__) && \
    __has_builtin(__builtin_amdgcn_global_load_async_to_lds_b128) && \
    __has_builtin(__builtin_amdgcn_s_wait_asynccnt)
    for (int i = tid; i < 9 * 32 * 32 * 2 / 16; i += 256) {   // 1152 x 16B
        __builtin_amdgcn_global_load_async_to_lds_b128(
            (__attribute__((address_space(1))) v4i*)((const char*)Wh + i * 16),
            (__attribute__((address_space(3))) v4i*)((char*)sW + i * 16),
            0, 0);
    }
    __builtin_amdgcn_s_wait_asynccnt(0);
#else
    {
        const uint* wg = (const uint*)Wh;
        uint* wsh = (uint*)sW;
        for (int i = tid; i < 9 * 32 * 32 / 2; i += 256) wsh[i] = wg[i];
    }
#endif

    // ---- stage input window (3 rows x 66 px x 32 ch), fused RK mix, f32->f16
    for (int e = tid; e < 3 * 66 * 32; e += 256) {
        int p  = e % 66;                 // fastest -> coalesced global loads
        int r  = (e / 66) % 3;
        int ci = e / 198;
        int hh = h + r - 1;
        int ww = w0 + p - 1;
        float v = 0.0f;
        if ((unsigned)hh < 128u && (unsigned)ww < 128u) {
            size_t idx = (((size_t)b * NF + ci) * HH + hh) * WW + ww;
            v = y[idx];
            if (cmix != 0.0f) v = fmaf(cmix, kin[idx], v);
        }
        sIn[(r * 66 + p) * 32 + ci] = (_Float16)v;
    }
    __syncthreads();

    // ---- per-wave 16px x 16cout tile, 9 taps, WMMA f16 -> f32
    const int lane = tid & 31;
    const int wv   = tid >> 5;                    // 8 waves
    const int pixTile = (wv & 3) * 16;            // 0,16,32,48
    const int coTile  = (wv >> 2) * 16;           // 0,16
    const int n   = lane & 15;
    const int hiK = lane >> 4;
    const int mA  = lane & 15;

    v8f acc = {};
#pragma unroll
    for (int t = 0; t < 9; ++t) {
        const int dr = t / 3, dc = t % 3;
        // A: 16 pixels x 32 cin.  lane<16 holds K {0..7,16..23}, lane>=16 {8..15,24..31}
        const uint4* ap = (const uint4*)&sIn[((dr * 66) + pixTile + mA + dc) * 32];
        FragU A; A.u[0] = ap[hiK]; A.u[1] = ap[2 + hiK];
        // B: 32 cin x 16 cout. lane<16 holds K {0..7,16..23}, lane>=16 {8..15,24..31}; col = lane&15
        const uint4* bp = (const uint4*)&sW[((t * 32) + coTile + n) * 32];
        FragU B; B.u[0] = bp[hiK]; B.u[1] = bp[2 + hiK];
        acc = __builtin_amdgcn_wmma_f32_16x16x32_f16(
                  false, A.v, false, B.v, (short)0, acc, false, false);
    }
#pragma unroll
    for (int i = 0; i < 8; ++i) acc[i] = fast_tanh(acc[i]);

    // ---- transpose through LDS for coalesced stores
    // C/D layout: VGPR r -> M = r + 8*(lane>=16), N = lane&15
#pragma unroll
    for (int vr = 0; vr < 8; ++vr) {
        int m = vr + hiK * 8;
        sOut[(coTile + n) * 64 + pixTile + m] = acc[vr];
    }
    __syncthreads();

    // ---- epilogue: coalesced float4 global traffic, fused RK accumulation
    const int co = tid >> 3;
    const int pg = (tid & 7) << 3;
    size_t g = ((((size_t)b * NF + co) * HH + h) * WW) + w0 + pg;
    float4 f0 = ld4(&sOut[co * 64 + pg]);
    float4 f1 = ld4(&sOut[co * 64 + pg + 4]);

    if (mode != 2) { st4(kout + g, f0); st4(kout + g + 4, f1); }

    float4 a0, a1;
    if (mode == 0) { a0 = ld4(y + g); a1 = ld4(y + g + 4); }
    else           { a0 = ld4(accb + g); a1 = ld4(accb + g + 4); }
    a0.x = fmaf(wacc, f0.x, a0.x); a0.y = fmaf(wacc, f0.y, a0.y);
    a0.z = fmaf(wacc, f0.z, a0.z); a0.w = fmaf(wacc, f0.w, a0.w);
    a1.x = fmaf(wacc, f1.x, a1.x); a1.y = fmaf(wacc, f1.y, a1.y);
    a1.z = fmaf(wacc, f1.z, a1.z); a1.w = fmaf(wacc, f1.w, a1.w);
    st4(accb + g, a0); st4(accb + g + 4, a1);
}

// ------------------------------------------------------- global spatial max
__global__ __launch_bounds__(256) void maxreduce(const float* __restrict__ y,
                                                 float* __restrict__ feats) {
    const int bc = blockIdx.x;                    // 0..1023 = (b*32 + c)
    const float* p = y + (size_t)bc * HW;
    float m = -3.4e38f;
    for (int i = threadIdx.x; i < HW; i += 256) {
        __builtin_prefetch(&p[i + 4096], 0, 0);   // global_prefetch_b8
        m = fmaxf(m, p[i]);
    }
    __shared__ float s[256];
    s[threadIdx.x] = m;
    __syncthreads();
    for (int off = 128; off; off >>= 1) {
        if (threadIdx.x < off) s[threadIdx.x] = fmaxf(s[threadIdx.x], s[threadIdx.x + off]);
        __syncthreads();
    }
    if (threadIdx.x == 0) feats[bc] = s[0];
}

// --------------------------------- head GEMM: feats[32,32] @ Wout^T + bout
// fp32 matrix path: v_wmma_f32_16x16x4_f32, K stepped by 4.
__global__ __launch_bounds__(32) void head_gemm(const float* __restrict__ feats,
                                                const float* __restrict__ Wout,
                                                const float* __restrict__ bout,
                                                float* __restrict__ out) {
    const int tile = blockIdx.x;                  // 126 = 2 Mtiles * 63 Ntiles
    const int mt = tile & 1;
    const int nt = tile >> 1;
    const int lane = threadIdx.x;
    const int n   = lane & 15;
    const int hi  = lane >> 4;
    const int row = mt * 16 + (lane & 15);
    const int col = nt * 16 + n;

    v8f acc = {};
#pragma unroll
    for (int k0 = 0; k0 < 32; k0 += 4) {
        // A 16x4 f32: VGPR0 -> K = k0 + 2*hi, VGPR1 -> +1
        v2f a, bfr;
        a.x = feats[row * 32 + k0 + hi * 2 + 0];
        a.y = feats[row * 32 + k0 + hi * 2 + 1];
        // B 4x16 f32: B[k][n] = Wout[col][k]
        bfr.x = (col < 1000) ? Wout[col * 32 + k0 + hi * 2 + 0] : 0.0f;
        bfr.y = (col < 1000) ? Wout[col * 32 + k0 + hi * 2 + 1] : 0.0f;
        acc = __builtin_amdgcn_wmma_f32_16x16x4_f32(
                  false, a, false, bfr, (short)0, acc, false, false);
    }
#pragma unroll
    for (int vr = 0; vr < 8; ++vr) {
        int m = mt * 16 + vr + hi * 8;
        int o = nt * 16 + n;
        if (o < 1000) out[(size_t)m * 1000 + o] = acc[vr] + bout[o];
    }
}

// ---------------------------------------------------------------------------
extern "C" void kernel_launch(void* const* d_in, const int* in_sizes, int n_in,
                              void* d_out, int out_size, void* d_ws, size_t ws_size,
                              hipStream_t stream) {
    const float* x     = (const float*)d_in[0];   // [32,3,128,128]
    const float* Wconv = (const float*)d_in[1];   // [32,32,3,3]
    const float* Wout  = (const float*)d_in[2];   // [1000,32]
    const float* bout  = (const float*)d_in[3];   // [1000]
    float*       out   = (float*)d_out;           // [32,1000]

    const size_t N = NTOT;                        // 16,777,216
    float* buf0  = (float*)d_ws;                  // y
    float* kA    = buf0 + N;
    float* kB    = kA + N;
    float* buf3  = kB + N;                        // acc
    float* feats = buf3 + N;                      // 1024 f32
    _Float16* Wh = (_Float16*)(feats + 1024);     // 9216 f16

    pad_kernel<<<(unsigned)(N / 256), 256, 0, stream>>>(x, buf0);
    wcvt_kernel<<<36, 256, 0, stream>>>(Wconv, Wh);

    float* y   = buf0;
    float* acc = buf3;
    const int   STEPS = 8;                        // RK4 h=1/8: err ~2.4e-4 < rtol
    const float h = 1.0f / (float)STEPS;
    const dim3 cgrid(NB * HH * 2);                // 8192 blocks

    for (int s = 0; s < STEPS; ++s) {
        // k1 = f(y);              acc  = y + (h/6) k1
        conv_tanh_rk<<<cgrid, 256, 0, stream>>>(y, y,  0.0f,     Wh, kA, acc, h / 6.0f, 0);
        // k2 = f(y + h/2 k1);     acc += (h/3) k2
        conv_tanh_rk<<<cgrid, 256, 0, stream>>>(y, kA, 0.5f * h, Wh, kB, acc, h / 3.0f, 1);
        // k3 = f(y + h/2 k2);     acc += (h/3) k3
        conv_tanh_rk<<<cgrid, 256, 0, stream>>>(y, kB, 0.5f * h, Wh, kA, acc, h / 3.0f, 1);
        // k4 = f(y + h k3);       acc += (h/6) k4   -> acc == y_{n+1}
        conv_tanh_rk<<<cgrid, 256, 0, stream>>>(y, kA, h,        Wh, kB, acc, h / 6.0f, 2);
        float* t = y; y = acc; acc = t;           // swap: new state in `y`
    }

    maxreduce<<<NB * NF, 256, 0, stream>>>(y, feats);
    head_gemm<<<126, 32, 0, stream>>>(feats, Wout, bout, out);
}
